// EntityTable_369367187856
// MI455X (gfx1250) — matrix operand order repaired and checked by reading, not compiled, not run on hardware
//
#include <hip/hip_runtime.h>
#include <hip/hip_bf16.h>

typedef __attribute__((ext_vector_type(16))) _Float16 v16h;
typedef __attribute__((ext_vector_type(8)))  float    v8f;
typedef __attribute__((ext_vector_type(4)))  float    v4f;

#define NB   16
#define NT   2048
#define ND   1024
#define NE   8
#define DE   64
#define G3   192              // 3*DE gate width
#define NCOL 200              // 192 G cols + 8 logit cols
#define NTIL 13               // ceil(208/16) N tiles in big GEMM
#define KTIL 32               // 1024/32 K tiles

// ---- workspace layout (bytes) ----
#define WS_WCFRAG 0                                  // 416*32*16 halves = 425984 B
#define WS_BIASG  425984                             // 192 f32
#define WS_LOGITS 430080                             // 32768*8 f32 = 1 MB
#define WS_ROUTE  1478656                            // 32768*8 f32 = 1 MB
#define WS_GBUF   2527232                            // 32768*192 f32 = 25.2 MB

// A-fragment index helpers (16-bit A 16x32, ISA 7.12.2)
__device__ __forceinline__ int a_lane(int m, int kk) { return m + ((kk & 8) << 1); }
__device__ __forceinline__ int a_half(int kk)        { return (kk & 7) | ((kk >> 1) & 8); }

// ---------------------------------------------------------------------------
// Kernel A: build Wc WMMA B-fragments.
// Wc[k][g] = sum_e Wi[e][k]*W_ih[g][e]   (g<192, combined proj*gate weight)
//          = entity_keys[g-192][k]       (192<=g<200, scaled at epilogue)
// B-frag (32x16 f16): lane%16 = n-in-tile, half s -> k = 8*(lane/16) + (s&7) + 16*(s>=8).
// ---------------------------------------------------------------------------
__global__ void et_prep_wc(const float* __restrict__ Wi,
                           const float* __restrict__ ekeys,
                           const float* __restrict__ W_ih,
                           _Float16* __restrict__ wcfrag) {
  const int kc   = blockIdx.x / NTIL;      // 0..31
  const int nt   = blockIdx.x % NTIL;      // 0..12
  const int lane = threadIdx.x;            // 0..31
  const int k0   = kc * 32 + ((lane >> 4) << 3);
  v16h bv;
  #pragma unroll
  for (int h = 0; h < 16; ++h) {
    const int g = nt * 16 + (lane & 15);
    const int k = k0 + (h & 7) + ((h & 8) << 1);
    float v = 0.0f;
    if (g < G3) {
      #pragma unroll 8
      for (int e = 0; e < DE; ++e) v += Wi[e * ND + k] * W_ih[g * DE + e];
    } else if (g < NCOL) {
      v = ekeys[(g - G3) * ND + k];
    }
    bv[h] = (_Float16)v;
  }
  ((v16h*)wcfrag)[blockIdx.x * 32 + lane] = bv;
}

// biasG[g] = dot(bi, W_ih[g,:])  (folds input-proj bias through W_ih)
__global__ void et_prep_bias(const float* __restrict__ bi,
                             const float* __restrict__ W_ih,
                             float* __restrict__ biasG) {
  const int g = threadIdx.x;               // 192 threads
  float v = 0.0f;
  for (int e = 0; e < DE; ++e) v += bi[e] * W_ih[g * DE + e];
  biasG[g] = v;
}

// ---------------------------------------------------------------------------
// Kernel B: big GEMM  X(32768x1024) x Wc(1024x200) via f16 WMMA.
// 2048 blocks x 13 waves; wave = N tile, block = 16-row M tile.
// A fragments built straight from global f32 rows: per lane the 16 halves are
// two contiguous 8-float K-runs (koff and koff+16), so 4x global_load_b128 +
// converts. No LDS, no barriers -> k-loop pipelines freely.
// ---------------------------------------------------------------------------
__global__ void et_gemm(const float* __restrict__ h_seq,
                        const _Float16* __restrict__ wcfrag,
                        const float* __restrict__ biasG,
                        float* __restrict__ Gbuf,
                        float* __restrict__ logits) {
  const int tid  = threadIdx.x;            // 0..415
  const int lane = tid & 31;
  const int wave = tid >> 5;               // 0..12 = N tile
  const size_t rowBase = (size_t)blockIdx.x * 16;

  const int mrow = lane & 15;              // A-matrix row for this lane
  const int koff = (lane >> 4) << 3;       // 0 (lanes 0-15) or 8 (lanes 16-31)
  const float* __restrict__ arow = h_seq + (rowBase + mrow) * ND + koff;

  v8f acc = {};
  #pragma unroll 4
  for (int kc = 0; kc < KTIL; ++kc) {
    const float* ap = arow + kc * 32;
    const v4f c0 = *(const v4f*)(ap + 0);
    const v4f c1 = *(const v4f*)(ap + 4);
    const v4f c2 = *(const v4f*)(ap + 16);
    const v4f c3 = *(const v4f*)(ap + 20);
    v16h a;
    #pragma unroll
    for (int j = 0; j < 4; ++j) {
      a[j]      = (_Float16)c0[j];
      a[4 + j]  = (_Float16)c1[j];
      a[8 + j]  = (_Float16)c2[j];
      a[12 + j] = (_Float16)c3[j];
    }
    const v16h bf = ((const v16h*)wcfrag)[(kc * NTIL + wave) * 32 + lane];
    acc = __builtin_amdgcn_wmma_f32_16x16x32_f16(false, a, false, bf,
                                                 (short)0, acc, false, false);
  }
  // epilogue: C/D layout — lane<16: M=r,N=lane; lane>=16: M=r+8,N=lane-16
  const int n0    = lane & 15;
  const int col   = wave * 16 + n0;
  const int mbase = (lane < 16) ? 0 : 8;
  const float bg  = (col < G3) ? biasG[col] : 0.0f;
  #pragma unroll
  for (int r = 0; r < 8; ++r) {
    const size_t row = rowBase + mbase + r;
    const float v = acc[r];
    if (col < G3)        Gbuf[row * G3 + col] = v + bg;
    else if (col < NCOL) logits[row * NE + (col - G3)] = v * 0.03125f; // 1/sqrt(1024)
  }
}

// ---------------------------------------------------------------------------
// Kernel C: softmax over N_E=8 per (b,t) row.
// ---------------------------------------------------------------------------
__global__ void et_softmax(const float* __restrict__ logits,
                           float* __restrict__ route) {
  const size_t row = (size_t)blockIdx.x * blockDim.x + threadIdx.x; // 32768 rows
  float l[NE];
  float mx = -3.0e38f;
  #pragma unroll
  for (int i = 0; i < NE; ++i) { l[i] = logits[row * NE + i]; mx = fmaxf(mx, l[i]); }
  float s = 0.0f;
  #pragma unroll
  for (int i = 0; i < NE; ++i) { l[i] = __expf(l[i] - mx); s += l[i]; }
  const float inv = 1.0f / s;
  #pragma unroll
  for (int i = 0; i < NE; ++i) route[row * NE + i] = l[i] * inv;
}

// ---------------------------------------------------------------------------
// Kernel D: sequential GRU scan. One block per batch, 4 waves.
// Per step: gh = state(8x64) @ W_hh^T via 6 WMMAs/wave; gx = w*G + b_ih.
// G rows and routing weights for step t+1 are register-prefetched during t.
// ---------------------------------------------------------------------------
__global__ void et_scan(const float* __restrict__ Gbuf,
                        const float* __restrict__ route,
                        const float* __restrict__ W_hh,
                        const float* __restrict__ b_ih,
                        const float* __restrict__ b_hh,
                        const float* __restrict__ e0,
                        float* __restrict__ out) {
  __shared__ v16h aFv[2 * 32];                 // state f16, A-frag layout (2 K-chunks)
  __shared__ __align__(16) float gh[16 * G3];  // gh + b_hh (rows 8..15 dead)
  _Float16* aFh = (_Float16*)aFv;

  const int tid  = threadIdx.x;            // 0..127
  const int lane = tid & 31;
  const int wave = tid >> 5;               // 0..3 -> N tiles 3w..3w+2
  const int b    = blockIdx.x;             // batch

  const int m  = tid >> 4;                 // entity row 0..7 (gate phase)
  const int d0 = (tid & 15) * 4;           // 4 consecutive d per thread

  // hoist W_hh B-fragments into registers (3 ntiles x 2 kchunks per wave)
  v16h bh[3][2];
  float bhh[3];
  #pragma unroll
  for (int j = 0; j < 3; ++j) {
    const int nt = wave * 3 + j;
    #pragma unroll
    for (int kc = 0; kc < 2; ++kc) {
      const int k = kc * 32 + ((lane >> 4) << 3);
      v16h bv;
      #pragma unroll
      for (int h = 0; h < 16; ++h) bv[h] = (_Float16)W_hh[(nt * 16 + (lane & 15)) * DE + k + (h & 7) + ((h & 8) << 1)];
      bh[j][kc] = bv;
    }
    bhh[j] = b_hh[nt * 16 + (lane & 15)];
  }

  // D-tile store coords (uniform, both lane halves write)
  const int dmbase = (lane < 16) ? 0 : 8;
  const int dn0    = lane & 15;

  // per-thread biases
  const v4f bir = *(const v4f*)&b_ih[d0];
  const v4f biz = *(const v4f*)&b_ih[DE + d0];
  const v4f bin = *(const v4f*)&b_ih[2 * DE + d0];

  // init state (rows 8..15 of A frag stay zero forever)
  for (int i = tid; i < 2 * 32 * 16; i += 128) aFh[i] = (_Float16)0.0f;
  __syncthreads();
  float st[4];
  #pragma unroll
  for (int j = 0; j < 4; ++j) {
    const int dg = d0 + j;
    const float v = e0[m * DE + dg];
    st[j] = v;
    aFh[(dg >> 5) * 512 + a_lane(m, dg & 31) * 16 + a_half(dg & 31)] = (_Float16)v;
  }
  __syncthreads();

  const size_t btBase = (size_t)b * NT;
  const size_t OUT2   = (size_t)NB * NT * (NE * DE);

  // software-pipelined prefetch of G row + routing weight
  v4f g_r = *(const v4f*)&Gbuf[btBase * G3 + d0];
  v4f g_z = *(const v4f*)&Gbuf[btBase * G3 + DE + d0];
  v4f g_n = *(const v4f*)&Gbuf[btBase * G3 + 2 * DE + d0];
  float wv = route[btBase * NE + m];

  for (int t = 0; t < NT; ++t) {
    // ---- WMMA phase: gh = state @ W_hh^T (+ b_hh) ----
    const v16h a0 = aFv[lane];
    const v16h a1 = aFv[32 + lane];
    #pragma unroll
    for (int j = 0; j < 3; ++j) {
      v8f acc = {};
      acc = __builtin_amdgcn_wmma_f32_16x16x32_f16(false, a0, false, bh[j][0],
                                                   (short)0, acc, false, false);
      acc = __builtin_amdgcn_wmma_f32_16x16x32_f16(false, a1, false, bh[j][1],
                                                   (short)0, acc, false, false);
      const int col = (wave * 3 + j) * 16 + dn0;
      #pragma unroll
      for (int r = 0; r < 8; ++r) gh[(dmbase + r) * G3 + col] = acc[r] + bhh[j];
    }
    __syncthreads();

    // ---- gate phase ----
    const v4f GR = g_r, GZ = g_z, GN = g_n;
    const float W = wv;
    if (t + 1 < NT) {                      // prefetch next step
      const float* gp = Gbuf + (btBase + t + 1) * G3;
      g_r = *(const v4f*)&gp[d0];
      g_z = *(const v4f*)&gp[DE + d0];
      g_n = *(const v4f*)&gp[2 * DE + d0];
      wv  = route[(btBase + t + 1) * NE + m];
    }
    const v4f ghr = *(const v4f*)&gh[m * G3 + d0];
    const v4f ghz = *(const v4f*)&gh[m * G3 + DE + d0];
    const v4f ghn = *(const v4f*)&gh[m * G3 + 2 * DE + d0];

    #pragma unroll
    for (int j = 0; j < 4; ++j) {
      const float r = 1.0f / (1.0f + __expf(-(W * GR[j] + bir[j] + ghr[j])));
      const float z = 1.0f / (1.0f + __expf(-(W * GZ[j] + biz[j] + ghz[j])));
      float x = W * GN[j] + bin[j] + r * ghn[j];
      x = fminf(fmaxf(x, -15.0f), 15.0f);
      const float e2 = __expf(2.0f * x);
      const float nn = (e2 - 1.0f) / (e2 + 1.0f);
      const float nv = (1.0f - z) * nn + z * st[j];
      st[j] = nv;

      const int dg = d0 + j;
      const size_t o1 = (btBase + t) * (NE * DE) + m * DE + dg;
      out[o1] = nv;                        // entity_seq view
      out[OUT2 + o1] = nv;                 // entity_stack view (same data)
      aFh[(dg >> 5) * 512 + a_lane(m, dg & 31) * 16 + a_half(dg & 31)] = (_Float16)nv;
    }
    __syncthreads();
  }
}

// ---------------------------------------------------------------------------
extern "C" void kernel_launch(void* const* d_in, const int* in_sizes, int n_in,
                              void* d_out, int out_size, void* d_ws, size_t ws_size,
                              hipStream_t stream) {
  const float* h_seq = (const float*)d_in[0];
  const float* ekeys = (const float*)d_in[1];
  const float* Wi    = (const float*)d_in[2];
  const float* bi    = (const float*)d_in[3];
  const float* W_ih  = (const float*)d_in[4];
  const float* W_hh  = (const float*)d_in[5];
  const float* b_ih  = (const float*)d_in[6];
  const float* b_hh  = (const float*)d_in[7];
  const float* e0    = (const float*)d_in[8];
  float* out = (float*)d_out;

  char* ws = (char*)d_ws;
  _Float16* wcfrag = (_Float16*)(ws + WS_WCFRAG);
  float* biasG  = (float*)(ws + WS_BIASG);
  float* logits = (float*)(ws + WS_LOGITS);
  float* routeW = (float*)(ws + WS_ROUTE);
  float* Gbuf   = (float*)(ws + WS_GBUF);

  et_prep_wc  <<<KTIL * NTIL, 32,  0, stream>>>(Wi, ekeys, W_ih, wcfrag);
  et_prep_bias<<<1, G3,            0, stream>>>(bi, W_ih, biasG);
  et_gemm     <<<(NB * NT) / 16, NTIL * 32, 0, stream>>>(h_seq, wcfrag, biasG,
                                                         Gbuf, logits);
  et_softmax  <<<(NB * NT) / 256, 256, 0, stream>>>(logits, routeW);
  et_scan     <<<NB, 128, 0, stream>>>(Gbuf, routeW, W_hh, b_ih, b_hh, e0, out);
}
